// FullAttention_15908558864468
// MI455X (gfx1250) — matrix operand order repaired and checked by reading, not compiled, not run on hardware
//
#include <hip/hip_runtime.h>

typedef __attribute__((ext_vector_type(16))) __bf16 bf16x16;
typedef __attribute__((ext_vector_type(8)))  __bf16 bf16x8;
typedef __attribute__((ext_vector_type(4)))  __bf16 bf16x4;
typedef __attribute__((ext_vector_type(8)))  float  f32x8;

#define B_ 4
#define L_ 2048
#define H_ 16
#define E_ 64

// scale (1/sqrt(64)) folded with log2(e) so we can use exp2 (v_exp_f32).
#define K_SCALE_LOG2E 0.18033688011112042f

// Load a 16-bit WMMA A-fragment from an LDS row.
// 16-bit A-layout (16x32), contiguous K order: lane holds row M=lane&15;
// K = 16*half + 0..15 (half = lane>>4), matching the B fragments.
// 'start' must already include 16*half (and any K-step offset).
__device__ __forceinline__ bf16x16 lds_a_frag(const __bf16* rowPtr, int start) {
  bf16x8 lo = *(const bf16x8*)(rowPtr + start);
  bf16x8 hi = *(const bf16x8*)(rowPtr + start + 8);
  bf16x16 a;
#pragma unroll
  for (int i = 0; i < 8; ++i) { a[i] = lo[i]; a[i + 8] = hi[i]; }
  return a;
}

__global__ __launch_bounds__(128)
void fa_fwd_causal(const float* __restrict__ Q, const float* __restrict__ K,
                   const float* __restrict__ V, float* __restrict__ O) {
  // K block (32 keys x 64 dims) row-major; V block transposed (64 dims x 32 keys).
  __shared__ __align__(16) __bf16 sK[32 * 64];
  __shared__ __align__(16) __bf16 sVT[64 * 32];

  const int qTiles = L_ / 64;
  const int blk = blockIdx.x;
  const int qt = blk % qTiles;
  const int bh = blk / qTiles;
  const int h = bh % H_;
  const int b = bh / H_;

  const int tid  = threadIdx.x;
  const int wave = tid >> 5;
  const int lane = tid & 31;
  const int col  = lane & 15;   // q column inside the tile (C-layout N)
  const int half = lane >> 4;   // lane half (C-layout M offset 0/8)

  const int qBase = qt * 64;
  const int q0    = qBase + wave * 16;  // this wave's first query row
  const int qAbs  = q0 + col;           // per-lane absolute query index

  // ---- Q B-fragments (held for the whole kernel) --------------------------
  // B fragment (32x16, 16-bit), this code's contiguous K order: lane n=lane&15 is column n; lanes 0-15 hold
  // K=0..15, lanes 16-31 hold K=16..31, sequential pairs per reg.
  // B[k][n] = Q[q0+n][es*32 + 16*half + k'] -> 16 contiguous floats per lane.
  bf16x16 qb[2];
  {
    const float* qrow = Q + ((size_t)(b * L_ + qAbs) * H_ + h) * E_;
#pragma unroll
    for (int f = 0; f < 2; ++f) {
      const float* p = qrow + f * 32 + 16 * half;
#pragma unroll
      for (int c4 = 0; c4 < 4; ++c4) {
        float4 x = *(const float4*)(p + 4 * c4);
        qb[f][4 * c4 + 0] = (__bf16)x.x;
        qb[f][4 * c4 + 1] = (__bf16)x.y;
        qb[f][4 * c4 + 2] = (__bf16)x.z;
        qb[f][4 * c4 + 3] = (__bf16)x.w;
      }
    }
  }

  // ---- staging addressing: ONE 64-bit pointer per tensor per thread ------
  // The 4 chunks a thread stages differ by a constant 8192 floats (32 KB) in
  // global memory and constant LDS strides, so they become immediate offsets
  // on global_load_b128 / ds_store; only the base pointers advance per block.
  const size_t planeOff = ((size_t)b * L_ * H_ + h) * E_;
  const int linear0 = tid * 4;          // this thread's first chunk
  const int sT = linear0 >> 6;          // key row 0..7
  const int eT = linear0 & 63;          // dim 0..60 (step 4)

  const float* kp = K + planeOff + (size_t)(sT * (H_ * E_) + eT);
  const float* vp = V + planeOff + (size_t)(sT * (H_ * E_) + eT);
  __bf16* skp  = sK  + sT * 64 + eT;    // sK[s][e]
  __bf16* svtp = sVT + eT * 32 + sT;    // sVT[e][s]

  // O^T accumulators: four 16x16 f32 C tiles (d-chunks of 16).
  f32x8 acc[4];
#pragma unroll
  for (int dc = 0; dc < 4; ++dc)
#pragma unroll
    for (int v = 0; v < 8; ++v) acc[dc][v] = 0.0f;

  float mI = -3.0e38f;  // running row max (per-lane q)
  float lI = 0.0f;      // running row sum

  const int nS = qBase / 32 + 2;  // key blocks covering s <= qBase+63

  for (int sb = 0; sb < nS; ++sb) {
    const int s0 = sb * 32;

    // ---- cooperative staging: K block and V^T block as bf16 in LDS -------
#pragma unroll
    for (int i = 0; i < 4; ++i) {
      float4 kv = *(const float4*)(kp + 8192 * i);  // imm offset 32768*i bytes
      bf16x4 pk;
      pk[0] = (__bf16)kv.x; pk[1] = (__bf16)kv.y;
      pk[2] = (__bf16)kv.z; pk[3] = (__bf16)kv.w;
      *(bf16x4*)(skp + 512 * i) = pk;               // imm offset 1024*i bytes

      float4 vv = *(const float4*)(vp + 8192 * i);
      svtp[8 * i]      = (__bf16)vv.x;              // imm offsets, 16-bit range
      svtp[8 * i + 32] = (__bf16)vv.y;
      svtp[8 * i + 64] = (__bf16)vv.z;
      svtp[8 * i + 96] = (__bf16)vv.w;
    }
    if (s0 + 32 < L_) {
      __builtin_prefetch(kp + 32 * (H_ * E_), 0, 0);
      __builtin_prefetch(vp + 32 * (H_ * E_), 0, 0);
    }
    __syncthreads();

    if (s0 <= q0 + 15) {  // wave-uniform: block intersects wave's causal range
      // ---- S^T = K . Q^T : two 16x16 C tiles (s-chunks), K-dim 64 --------
      f32x8 st[2];
#pragma unroll
      for (int sc = 0; sc < 2; ++sc) {
        f32x8 c;
#pragma unroll
        for (int v = 0; v < 8; ++v) c[v] = 0.0f;
#pragma unroll
        for (int es = 0; es < 2; ++es) {
          bf16x16 aK = lds_a_frag(sK + (sc * 16 + col) * 64, es * 32 + 16 * half);
          c = __builtin_amdgcn_wmma_f32_16x16x32_bf16(false, aK, false, qb[es],
                                                      (short)0, c, false, false);
        }
        st[sc] = c;
      }

      // ---- causal mask (C-layout: M = s0 + sc*16 + 8*half + v, N = q) ----
      if (s0 + 31 > q0) {
#pragma unroll
        for (int v = 0; v < 8; ++v) {
          const int sA0 = s0 + 8 * half + v;
          st[0][v] = (sA0 <= qAbs) ? st[0][v] : -3.0e38f;
          st[1][v] = (sA0 + 16 <= qAbs) ? st[1][v] : -3.0e38f;
        }
      }

      // ---- online softmax: stats per lane (per q), one cross-half shuffle
      float mloc = st[0][0];
#pragma unroll
      for (int v = 1; v < 8; ++v) mloc = fmaxf(mloc, st[0][v]);
#pragma unroll
      for (int v = 0; v < 8; ++v) mloc = fmaxf(mloc, st[1][v]);
      mloc = fmaxf(mloc, __shfl_xor(mloc, 16, 32));

      const float mnew = fmaxf(mI, mloc);
      const float fac  = exp2f((mI - mnew) * K_SCALE_LOG2E);

      float p0[8], p1[8];
      float ls = 0.0f;
#pragma unroll
      for (int v = 0; v < 8; ++v) {
        p0[v] = exp2f((st[0][v] - mnew) * K_SCALE_LOG2E);
        p1[v] = exp2f((st[1][v] - mnew) * K_SCALE_LOG2E);
        ls += p0[v] + p1[v];
      }
      ls += __shfl_xor(ls, 16, 32);
      lI = lI * fac + ls;
      mI = mnew;

#pragma unroll
      for (int dc = 0; dc < 4; ++dc)
#pragma unroll
        for (int v = 0; v < 8; ++v) acc[dc][v] *= fac;

      // ---- repack P^T (C-layout) into B-fragment: one shuffle, no LDS ----
      // low lane needs s=0..7 (local p0) + s=8..15 (partner p0);
      // high lane needs s=16..23 (partner p1) + s=24..31 (local p1).
      bf16x16 pb;
#pragma unroll
      for (int v = 0; v < 8; ++v) {
        const float sel = half ? p0[v] : p1[v];
        const float rec = __shfl_xor(sel, 16, 32);
        pb[v]     = (__bf16)(half ? rec   : p0[v]);
        pb[v + 8] = (__bf16)(half ? p1[v] : rec);
      }

      // ---- O^T += V^T . P^T : four d-chunks, K-dim 32 --------------------
#pragma unroll
      for (int dc = 0; dc < 4; ++dc) {
        bf16x16 aV = lds_a_frag(sVT + (dc * 16 + col) * 32, 16 * half);
        acc[dc] = __builtin_amdgcn_wmma_f32_16x16x32_bf16(false, aV, false, pb,
                                                          (short)0, acc[dc],
                                                          false, false);
      }
    }
    __syncthreads();

    kp += 32 * (H_ * E_);  // advance one key block: single 64-bit add each
    vp += 32 * (H_ * E_);
  }

  // ---- normalize and store: O[b][q][h][d], contiguous float4 per chunk ---
  const float inv = 1.0f / lI;
  float* orow = O + ((size_t)(b * L_ + qAbs) * H_ + h) * E_;
#pragma unroll
  for (int dc = 0; dc < 4; ++dc) {
    float4 o0, o1;
    o0.x = acc[dc][0] * inv; o0.y = acc[dc][1] * inv;
    o0.z = acc[dc][2] * inv; o0.w = acc[dc][3] * inv;
    o1.x = acc[dc][4] * inv; o1.y = acc[dc][5] * inv;
    o1.z = acc[dc][6] * inv; o1.w = acc[dc][7] * inv;
    *(float4*)(orow + dc * 16 + 8 * half)     = o0;
    *(float4*)(orow + dc * 16 + 8 * half + 4) = o1;
  }
}

extern "C" void kernel_launch(void* const* d_in, const int* in_sizes, int n_in,
                              void* d_out, int out_size, void* d_ws, size_t ws_size,
                              hipStream_t stream) {
  (void)in_sizes; (void)n_in; (void)out_size; (void)d_ws; (void)ws_size;
  const float* q = (const float*)d_in[0];
  const float* k = (const float*)d_in[1];
  const float* v = (const float*)d_in[2];
  float* o = (float*)d_out;

  dim3 grid(B_ * H_ * (L_ / 64));  // 2048 workgroups
  dim3 block(128);                 // 4 waves (wave32)
  hipLaunchKernelGGL(fa_fwd_causal, grid, block, 0, stream, q, k, v, o);
}
